// VarianceAdaptor_29437705847510
// MI455X (gfx1250) — matrix operand order repaired and checked by reading, not compiled, not run on hardware
//
#include <hip/hip_runtime.h>
#include <hip/hip_bf16.h>
#include <math.h>

// ---------------------------------------------------------------------------
// Problem constants: B=32, P(T)=1024, D=F=384, M=2048, NBINS=256, K=3, eps=1e-5
// ---------------------------------------------------------------------------
#define BB    32
#define TT    1024
#define DD    384
#define FF    384
#define MM    2048
#define NBINS 256
#define EPSB  1e-5f
#define CIN   384

typedef __bf16 v16bf __attribute__((ext_vector_type(16)));
typedef __bf16 v8bf  __attribute__((ext_vector_type(8)));
typedef __bf16 v4bf  __attribute__((ext_vector_type(4)));
typedef float  v8f   __attribute__((ext_vector_type(8)));

// LDS strides chosen so every fragment load is 16-byte aligned -> ds_load_b128
#define SA_STR 40   // halves per A row (80 B)
#define SB_STR 40   // halves per B row (80 B)
#define SA_SZ  (66 * SA_STR)
#define SB_SZ  (3 * 128 * SB_STR)

// ---------------------------------------------------------------------------
// Weight pre-pack: f32 [F][CIN][3] (taps innermost) -> bf16 [tap][F][CIN]
// so conv staging can use contiguous 128-bit loads.
// ---------------------------------------------------------------------------
__global__ __launch_bounds__(256)
void pack_weights(const float* __restrict__ w, __bf16* __restrict__ wpk) {
    const int idx = blockIdx.x * 256 + threadIdx.x;   // 3*FF*CIN total, exact
    const int tap = idx / (FF * CIN);
    const int rem = idx % (FF * CIN);
    const int f = rem / CIN, k = rem % CIN;
    wpk[idx] = (__bf16)w[((size_t)f * CIN + k) * 3 + tap];
}

// ---------------------------------------------------------------------------
// Staging helpers (vectorized, zero-padded halo rows for t=-1 / t=T)
// ---------------------------------------------------------------------------
__device__ __forceinline__ void stage_A(__bf16* sA, const float* in,
                                        int b, int t0, int k0) {
    for (int idx = threadIdx.x; idx < 66 * 8; idx += 256) {
        const int r = idx >> 3, c4 = (idx & 7) * 4;
        const int gt = t0 - 1 + r;
        v4bf o;
        if (gt >= 0 && gt < TT) {
            const float4 v =
                *(const float4*)&in[((size_t)b * TT + gt) * CIN + k0 + c4];
            o[0] = (__bf16)v.x; o[1] = (__bf16)v.y;
            o[2] = (__bf16)v.z; o[3] = (__bf16)v.w;
        } else {
            o[0] = o[1] = o[2] = o[3] = (__bf16)0.0f;
        }
        *(v4bf*)&sA[r * SA_STR + c4] = o;
    }
}

__device__ __forceinline__ void stage_A(__bf16* sA, const __bf16* in,
                                        int b, int t0, int k0) {
    for (int idx = threadIdx.x; idx < 66 * 4; idx += 256) {
        const int r = idx >> 2, c8 = (idx & 3) * 8;
        const int gt = t0 - 1 + r;
        v8bf v;
        if (gt >= 0 && gt < TT) {
            v = *(const v8bf*)&in[((size_t)b * TT + gt) * CIN + k0 + c8];
        } else {
#pragma unroll
            for (int e = 0; e < 8; ++e) v[e] = (__bf16)0.0f;
        }
        *(v8bf*)&sA[r * SA_STR + c8] = v;
    }
}

__device__ __forceinline__ void stage_B(__bf16* sB, const __bf16* wpk,
                                        int f0, int k0) {
    for (int idx = threadIdx.x; idx < 3 * 128 * 4; idx += 256) {
        const int tap = idx >> 9;            // / 512
        const int rem = idx & 511;
        const int f = rem >> 2, c8 = (rem & 3) * 8;
        const v8bf v =
            *(const v8bf*)&wpk[((size_t)tap * FF + f0 + f) * CIN + k0 + c8];
        *(v8bf*)&sB[(tap * 128 + f) * SB_STR + c8] = v;
    }
}

// ---------------------------------------------------------------------------
// Fused conv1d(K=3, SAME) + bias + inference-BN + ReLU via bf16 WMMA.
//   in  : [B, T, CIN]  (float layer1 / bf16 layer2)
//   wpk : [3, F, CIN]  bf16 (pre-packed)
//   out : [B, T, F]    bf16
// Block tile 64(T) x 128(F); 8 waves; wave = (tm = wv&3, tn0 = (wv>>2)*4).
// Each wave: 4 accumulators; per K-step 3 taps x 4 = 12 WMMAs; A fragment
// reused 4x. Double-buffered LDS hides staging behind the matrix pipe.
// ---------------------------------------------------------------------------
template <typename TIN>
__global__ __launch_bounds__(256)
void conv_bn_relu_wmma(const TIN* __restrict__ in,
                       const __bf16* __restrict__ wpk,
                       const float* __restrict__ bias,
                       const float* __restrict__ gamma,
                       const float* __restrict__ beta,
                       __bf16* __restrict__ out) {
    __shared__ __align__(16) __bf16 sA[2 * SA_SZ];
    __shared__ __align__(16) __bf16 sB[2 * SB_SZ];

    const int b  = blockIdx.x / (TT / 64);
    const int t0 = (blockIdx.x % (TT / 64)) * 64;
    const int f0 = blockIdx.y * 128;

    const int lane = threadIdx.x & 31;
    const int wv   = threadIdx.x >> 5;
    const int tm   = wv & 3;              // T-subtile (0..3)
    const int tn0  = (wv >> 2) * 4;       // first of four F-subtiles

    const int nrow = lane & 15;           // A-row m / B-col n / C-col n
    const int kh   = lane >> 4;           // K-half select

    v8f acc[4] = {};

    constexpr int NK = CIN / 32;          // 12 K-steps
    stage_A(sA, in, b, t0, 0);
    stage_B(sB, wpk, f0, 0);

    int buf = 0;
    for (int ks = 0; ks < NK; ++ks) {
        __syncthreads();                  // staged(buf) ready; buf^1 free
        if (ks + 1 < NK) {
            stage_A(sA + (buf ^ 1) * SA_SZ, in, b, t0, (ks + 1) * 32);
            stage_B(sB + (buf ^ 1) * SB_SZ, wpk, f0, (ks + 1) * 32);
        }
        if (ks + 2 < NK)                  // warm L2/WGP$ for the step after
            __builtin_prefetch(
                (const void*)((const char*)in +
                              (((size_t)b * TT + t0) * CIN + (ks + 2) * 32) *
                                  sizeof(TIN)), 0, 0);

        const __bf16* cA = sA + buf * SA_SZ;
        const __bf16* cB = sB + buf * SB_SZ;
#pragma unroll
        for (int tap = 0; tap < 3; ++tap) {
            // A 16x32 fragment: lane holds row m = lane&15;
            // lanes 0-15 K {0..7,16..23}; lanes 16-31 K {8..15,24..31}.
            const __bf16* arow = cA + (tm * 16 + nrow + tap) * SA_STR;
            const v8bf a0 = *(const v8bf*)(arow + kh * 8);
            const v8bf a1 = *(const v8bf*)(arow + 16 + kh * 8);
            const v16bf a = __builtin_shufflevector(
                a0, a1, 0, 1, 2, 3, 4, 5, 6, 7, 8, 9, 10, 11, 12, 13, 14, 15);
#pragma unroll
            for (int j = 0; j < 4; ++j) {
                // B 32x16 fragment: lane holds col n = lane&15;
                // lanes 0-15 K {0..7,16..23}; lanes 16-31 K {8..15,24..31}.
                const __bf16* brow =
                    cB + (tap * 128 + (tn0 + j) * 16 + nrow) * SB_STR + kh * 8;
                const v8bf b0 = *(const v8bf*)(brow);
                const v8bf b1 = *(const v8bf*)(brow + 16);
                const v16bf bv = __builtin_shufflevector(
                    b0, b1, 0, 1, 2, 3, 4, 5, 6, 7, 8, 9, 10, 11, 12, 13, 14,
                    15);
                acc[j] = __builtin_amdgcn_wmma_f32_16x16x32_bf16(
                    false, a, false, bv, (short)0, acc[j], false, false);
            }
        }
        buf ^= 1;
    }

    // ---- epilogue: (+bias) * g/sqrt(1+eps) + beta, ReLU, store bf16 -----
    const float rs = rsqrtf(1.0f + EPSB);
#pragma unroll
    for (int j = 0; j < 4; ++j) {
        const int f = f0 + (tn0 + j) * 16 + nrow;
        const float sc = gamma[f] * rs;
        const float bi = bias[f];
        const float be = beta[f];
#pragma unroll
        for (int v = 0; v < 8; ++v) {
            const int mmv = v + 8 * kh;          // C layout: M = vgpr + 8*half
            const int t   = t0 + tm * 16 + mmv;
            float val = (acc[j][v] + bi) * sc + be;
            val = val > 0.0f ? val : 0.0f;
            out[((size_t)b * TT + t) * FF + f] = (__bf16)val;
        }
    }
}

// ---------------------------------------------------------------------------
// Projection F->1: o[row] = sum_f h[row,f]*pw[f] + pb ; mode=1 => exp(o)-1.
// ---------------------------------------------------------------------------
__global__ __launch_bounds__(256)
void proj_kernel(const __bf16* __restrict__ h, const float* __restrict__ pw,
                 const float* __restrict__ pb, float* __restrict__ o, int mode) {
    const int lane = threadIdx.x & 31;
    const int row  = blockIdx.x * 8 + (threadIdx.x >> 5);
    const __bf16* hr = h + (size_t)row * FF;
    float s = 0.0f;
#pragma unroll
    for (int i = 0; i < FF / 32; ++i) {
        const int f = lane + i * 32;
        s += (float)hr[f] * pw[f];
    }
#pragma unroll
    for (int off = 16; off > 0; off >>= 1) s += __shfl_xor(s, off, 32);
    if (lane == 0) {
        float v = s + pb[0];
        if (mode) v = expf(v) - 1.0f;
        o[row] = v;
    }
}

// ---------------------------------------------------------------------------
// Per-batch inclusive cumsum of durations (P=1024) + mel_lens output.
// ---------------------------------------------------------------------------
__global__ __launch_bounds__(256)
void cumsum_kernel(const int* __restrict__ dur, int* __restrict__ ends,
                   float* __restrict__ lens_out) {
    __shared__ int sc[256];
    const int b   = blockIdx.x;
    const int tid = threadIdx.x;
    const int base = tid * 4;
    int d0 = dur[b * TT + base + 0];
    int d1 = dur[b * TT + base + 1];
    int d2 = dur[b * TT + base + 2];
    int d3 = dur[b * TT + base + 3];
    int p0 = d0, p1 = p0 + d1, p2 = p1 + d2, p3 = p2 + d3;
    sc[tid] = p3;
    __syncthreads();
    for (int off = 1; off < 256; off <<= 1) {
        int v = (tid >= off) ? sc[tid - off] : 0;
        __syncthreads();
        sc[tid] += v;
        __syncthreads();
    }
    const int excl = sc[tid] - p3;
    ends[b * TT + base + 0] = excl + p0;
    ends[b * TT + base + 1] = excl + p1;
    ends[b * TT + base + 2] = excl + p2;
    ends[b * TT + base + 3] = excl + p3;
    if (tid == 255) {
        int total = sc[255];
        lens_out[b] = (float)(total < MM ? total : MM);
    }
}

__device__ __forceinline__ int lower_bound_bins(const float* bins, float v) {
    int lo = 0, hi = NBINS - 1;           // 255 bin edges
    while (lo < hi) {
        int mid = (lo + hi) >> 1;
        if (bins[mid] < v) lo = mid + 1; else hi = mid;
    }
    return lo;
}

// ---------------------------------------------------------------------------
// Length regulation + pitch/energy embedding add + mel mask.
// ---------------------------------------------------------------------------
__global__ __launch_bounds__(384)
void expand_kernel(const float* __restrict__ x, const int* __restrict__ ends,
                   const float* __restrict__ pitch, const float* __restrict__ energy,
                   const float* __restrict__ pbins, const float* __restrict__ ebins,
                   const float* __restrict__ pemb, const float* __restrict__ eemb,
                   float* __restrict__ out_x, float* __restrict__ out_mask) {
    __shared__ int sidx[4], svalid[4], spb[4], seb[4];
    const int b = blockIdx.y;
    const int tid = threadIdx.x;
    if (tid < 4) {
        const int t = blockIdx.x * 4 + tid;
        const int* e = ends + b * TT;
        int total = e[TT - 1];
        int melLen = total < MM ? total : MM;
        int lo = 0, hi = TT;                  // upper_bound: first ends[i] > t
        while (lo < hi) {
            int mid = (lo + hi) >> 1;
            if (e[mid] <= t) lo = mid + 1; else hi = mid;
        }
        sidx[tid]   = lo < (TT - 1) ? lo : (TT - 1);
        svalid[tid] = (t < melLen) ? 1 : 0;
        spb[tid]    = lower_bound_bins(pbins, pitch[b * MM + t]);
        seb[tid]    = lower_bound_bins(ebins, energy[b * MM + t]);
        out_mask[b * MM + t] = svalid[tid] ? 0.0f : 1.0f;
    }
    __syncthreads();
#pragma unroll
    for (int fr = 0; fr < 4; ++fr) {
        const int t = blockIdx.x * 4 + fr;
        float base = svalid[fr]
            ? x[((size_t)b * TT + sidx[fr]) * DD + tid] : 0.0f;
        base += pemb[(size_t)spb[fr] * DD + tid];
        base += eemb[(size_t)seb[fr] * DD + tid];
        out_x[((size_t)b * MM + t) * DD + tid] = base;
    }
}

// ---------------------------------------------------------------------------
// Launch. Workspace: h1 (24MB) | h2 (24MB) | ends (128KB) | wpk (0.85MB)
// ---------------------------------------------------------------------------
extern "C" void kernel_launch(void* const* d_in, const int* in_sizes, int n_in,
                              void* d_out, int out_size, void* d_ws, size_t ws_size,
                              hipStream_t stream) {
    const float* x      = (const float*)d_in[0];
    const int*   dur    = (const int*)  d_in[3];
    const float* pitch  = (const float*)d_in[4];
    const float* energy = (const float*)d_in[5];
    const float* c1w    = (const float*)d_in[7];
    const float* c1b    = (const float*)d_in[8];
    const float* g1     = (const float*)d_in[9];
    const float* b1     = (const float*)d_in[10];
    const float* c2w    = (const float*)d_in[11];
    const float* c2b    = (const float*)d_in[12];
    const float* g2     = (const float*)d_in[13];
    const float* b2     = (const float*)d_in[14];
    const float* pw     = (const float*)d_in[15];
    const float* pbias  = (const float*)d_in[16];
    const float* pbins  = (const float*)d_in[17];
    const float* ebins  = (const float*)d_in[18];
    const float* pemb   = (const float*)d_in[19];
    const float* eemb   = (const float*)d_in[20];

    float* out = (float*)d_out;
    const size_t N_XEXP  = (size_t)BB * MM * DD;
    const size_t OFF_DUR = N_XEXP;
    const size_t OFF_PIT = OFF_DUR + (size_t)BB * TT;
    const size_t OFF_EN  = OFF_PIT + (size_t)BB * TT;
    const size_t OFF_LEN = OFF_EN  + (size_t)BB * TT;
    const size_t OFF_MSK = OFF_LEN + BB;

    char* ws = (char*)d_ws;
    const size_t HBYTES = (size_t)BB * TT * FF * sizeof(__bf16);   // 24 MB
    __bf16* h1  = (__bf16*)ws;
    __bf16* h2  = (__bf16*)(ws + HBYTES);
    int*   ends = (int*)  (ws + 2 * HBYTES);
    __bf16* wpk = (__bf16*)(ws + 2 * HBYTES + (size_t)BB * TT * sizeof(int));

    const dim3 gconv(BB * (TT / 64), FF / 128);
    const dim3 gpack((3 * FF * CIN) / 256);
    const dim3 blk(256);
    float* predOut[3] = {out + OFF_DUR, out + OFF_PIT, out + OFF_EN};

    for (int i = 0; i < 3; ++i) {
        pack_weights<<<gpack, blk, 0, stream>>>(c1w + (size_t)i * FF * DD * 3, wpk);
        conv_bn_relu_wmma<float><<<gconv, blk, 0, stream>>>(
            x, wpk, c1b + i * FF, g1 + i * FF, b1 + i * FF, h1);
        pack_weights<<<gpack, blk, 0, stream>>>(c2w + (size_t)i * FF * FF * 3, wpk);
        conv_bn_relu_wmma<__bf16><<<gconv, blk, 0, stream>>>(
            h1, wpk, c2b + i * FF, g2 + i * FF, b2 + i * FF, h2);
        proj_kernel<<<dim3((BB * TT) / 8), blk, 0, stream>>>(
            h2, pw + (size_t)i * FF, pbias + i, predOut[i], i == 0 ? 1 : 0);
    }

    cumsum_kernel<<<dim3(BB), dim3(256), 0, stream>>>(dur, ends, out + OFF_LEN);
    expand_kernel<<<dim3(MM / 4, BB), dim3(384), 0, stream>>>(
        x, ends, pitch, energy, pbins, ebins, pemb, eemb,
        out, out + OFF_MSK);
}
